// IDStructureStream_7438883357143
// MI455X (gfx1250) — compile-verified, matrix operand order checked by reading, not run on hardware
//
#include <hip/hip_runtime.h>

// ---------------------------------------------------------------------------
// VMamba block for MI455X (gfx1250): bf16 WMMA GEMMs + TDM/async-LDS staging
// + fused LN / scan / gate.
// ---------------------------------------------------------------------------

typedef __attribute__((ext_vector_type(16))) __bf16 v16bf;
typedef __attribute__((ext_vector_type(2)))  __bf16 v2bf;
typedef __attribute__((ext_vector_type(8)))  float  v8f;
typedef __attribute__((ext_vector_type(4)))  int    int4v;

#define BB     8
#define CDIM   256
#define HIN    96
#define WIN    96
#define HO     48
#define WO     48
#define LSEQ   2304          // 48*48
#define NPOS   18432         // BB*LSEQ
#define DIN    512
#define DSTATE 16
#define DTRANK 16
#define KDOWN  1024          // 256 * 2 * 2

#define AS1 __attribute__((address_space(1)))
#define AS3 __attribute__((address_space(3)))

// ---- feature detection (device pass only; host pass sees fallbacks) -------
#if defined(__HIP_DEVICE_COMPILE__)
# if __has_builtin(__builtin_amdgcn_tensor_load_to_lds) && \
     __has_builtin(__builtin_amdgcn_s_wait_tensorcnt) && (__clang_major__ < 23)
#  define USE_TDM 1          // probe-verified 5-arg arity on this lane
# else
#  define USE_TDM 0
# endif
# if __has_builtin(__builtin_amdgcn_global_load_async_to_lds_b128) && \
     __has_builtin(__builtin_amdgcn_global_load_async_to_lds_b32)  && \
     __has_builtin(__builtin_amdgcn_s_wait_asynccnt)
#  define USE_ASYNC 1
# else
#  define USE_ASYNC 0
# endif
#else
# define USE_TDM   0
# define USE_ASYNC 0
#endif

__device__ __forceinline__ __bf16 f2bf(float f) {
  union { float f; unsigned u; } v; v.f = f;
  unsigned r = (v.u + 0x7FFFu + ((v.u >> 16) & 1u)) >> 16;
  union { unsigned short s; __bf16 b; } o; o.s = (unsigned short)r;
  return o.b;
}

__device__ __forceinline__ v8f wmma_bf16(v16bf a, v16bf b, v8f c) {
  // (neg_a, A, neg_b, B, c_mod, C, reuse_a, reuse_b)
  return __builtin_amdgcn_wmma_f32_16x16x32_bf16(false, a, false, b,
                                                 (short)0, c, false, false);
}

// Load one 16x32 bf16 fragment row (row = lane&15, half = lane>>4) from a
// K-contiguous row.  ISA 16-bit layout: VGPR v<4 holds K=2v (+8 upper half);
// v>=4 holds K=16+2(v-4) (+8).  Pairs are K-contiguous -> 32-bit loads.
__device__ __forceinline__ v16bf load_frag_k(const __bf16* base, int half) {
  v16bf f;
#pragma unroll
  for (int v = 0; v < 8; ++v) {
    int k0 = ((v < 4) ? (v * 2) : (16 + (v - 4) * 2)) + half * 8;
    v2bf t = *(const v2bf*)(base + k0);
    f[2 * v]     = t.x;
    f[2 * v + 1] = t.y;
  }
  return f;
}

#if USE_ASYNC
// Builtins take typed pointers: (AS1 int4*/int* src, AS3 int4*/int* dst,
// imm offset, imm cpol) — confirmed via compiler diagnostics.
__device__ __forceinline__ void async_b128(const void* g, void* l) {
  __builtin_amdgcn_global_load_async_to_lds_b128((AS1 int4v*)g, (AS3 int4v*)l, 0, 0);
}
__device__ __forceinline__ void async_b32(const void* g, void* l) {
  __builtin_amdgcn_global_load_async_to_lds_b32((AS1 int*)g, (AS3 int*)l, 0, 0);
}
#endif

#if USE_TDM
typedef __attribute__((ext_vector_type(4))) unsigned uint4v;
typedef __attribute__((ext_vector_type(8))) int      int8v;

// One TDM descriptor moves a rows x kelems bf16 tile (rows contiguous, row
// stride kelems) into LDS, inserting 8 bf16 (4 DWORDs, pad code 3) after each
// row so the LDS image matches PITCH = kelems + 8.
__device__ __forceinline__ void tdm_load_panel(const void* gsrc, void* lds,
                                               unsigned kelems, unsigned rows,
                                               unsigned pad_interval_code) {
  unsigned long long ga = (unsigned long long)gsrc;
  unsigned la = (unsigned)(unsigned long long)(AS3 void*)lds;
  uint4v g0;
  g0.x = 1u;                                            // count=1 (valid D#)
  g0.y = la;                                            // lds_addr
  g0.z = (unsigned)ga;                                  // global_addr[31:0]
  g0.w = (unsigned)((ga >> 32) & 0x01ffffffu) | (2u << 30);  // addr[56:32]|type=2
  unsigned w0 = (1u << 16)                              // data_size = 2 bytes
              | (1u << 20)                              // pad_enable
              | (pad_interval_code << 22)               // row length in DWORDs
              | (3u << 25);                             // pad 4 DWORDs (8 bf16)
  unsigned w1 = (kelems & 0xffffu) << 16;               // tensor_dim0[15:0]
  unsigned w2 = (kelems >> 16) | ((rows & 0xffffu) << 16);   // td0 hi | td1 lo
  unsigned w3 = (rows >> 16) | ((kelems & 0xffffu) << 16);   // td1 hi | tile_dim0
  unsigned w4 = rows & 0xffffu;                         // tile_dim1 (tile_dim2=0)
  unsigned w5 = kelems;                                 // tensor_dim0_stride lo
  int8v g1 = { (int)w0, (int)w1, (int)w2, (int)w3,
               (int)w4, (int)w5, 0, 0 };
  int4v z4 = { 0, 0, 0, 0 };
  __builtin_amdgcn_tensor_load_to_lds(g0, g1, z4, z4, 0);
}
#endif

// Stage 16 contiguous rows of KELEMS bf16 into a padded LDS panel.
template <int KELEMS, int PAD_CODE>
__device__ __forceinline__ void stage_panel(const __bf16* grows, __bf16* panel,
                                            int tid) {
  const int PITCH = KELEMS + 8;
#if USE_TDM
  if (tid < 32) {                          // wave 0 drives the tensor DMA
    tdm_load_panel(grows, panel, (unsigned)KELEMS, 16u, (unsigned)PAD_CODE);
    __builtin_amdgcn_s_wait_tensorcnt(0);
  }
  __syncthreads();
#elif USE_ASYNC
  for (int e = tid; e < 16 * KELEMS / 8; e += 256) {
    int p = e / (KELEMS / 8), kk = e - p * (KELEMS / 8);
    async_b128(grows + p * KELEMS + kk * 8, panel + p * PITCH + kk * 8);
  }
  __builtin_amdgcn_s_wait_asynccnt(0);
  __syncthreads();
#else
  for (int e = tid; e < 16 * KELEMS; e += 256) {
    int p = e / KELEMS, k = e - p * KELEMS;
    panel[p * PITCH + k] = grows[p * KELEMS + k];
  }
  __syncthreads();
#endif
}

// ---------------------------------------------------------------------------
// K0a: fp32 -> bf16 weight conversion (row-major, K-contiguous)
// ---------------------------------------------------------------------------
__global__ __launch_bounds__(256) void k_convert(
    const float* __restrict__ wdown, const float* __restrict__ win,
    const float* __restrict__ wxproj, const float* __restrict__ wout,
    __bf16* __restrict__ wdown_bf, __bf16* __restrict__ win_bf,
    __bf16* __restrict__ wxproj_bf, __bf16* __restrict__ wout_bf) {
  const int N0 = CDIM * KDOWN;        // w_down [256][1024] (ci*4+dy*2+dx contig)
  const int N1 = 2 * DIN * CDIM;      // w_in   [1024][256]
  const int N2 = 48 * DIN;            // w_xproj[48][512]
  const int N3 = CDIM * DIN;          // w_out  [256][512]
  int idx = blockIdx.x * 256 + threadIdx.x;
  if (idx < N0)                       wdown_bf [idx]                = f2bf(wdown [idx]);
  else if (idx < N0 + N1)             win_bf   [idx - N0]           = f2bf(win   [idx - N0]);
  else if (idx < N0 + N1 + N2)        wxproj_bf[idx - N0 - N1]      = f2bf(wxproj[idx - N0 - N1]);
  else if (idx < N0 + N1 + N2 + N3)   wout_bf  [idx - N0 - N1 - N2] = f2bf(wout  [idx - N0 - N1 - N2]);
}

// ---------------------------------------------------------------------------
// K0b: x fp32 -> bf16 (enables 4-byte async gathers in K1)
// ---------------------------------------------------------------------------
__global__ __launch_bounds__(256) void k_convert_x(
    const float* __restrict__ x, __bf16* __restrict__ x_bf) {
  int idx = blockIdx.x * 256 + threadIdx.x;   // grid covers exactly BB*CDIM*HIN*WIN
  x_bf[idx] = f2bf(x[idx]);
}

// ---------------------------------------------------------------------------
// K1: patch-merge conv (implicit GEMM M=256,K=1024) + LN1 + LN2 -> xs (bf16)
// One block = 16 output positions; 8 waves x 2 M-tiles cover 256 channels.
// ---------------------------------------------------------------------------
__global__ __launch_bounds__(256) void k_merge_ln(
    const __bf16* __restrict__ x_bf, const __bf16* __restrict__ wdown_bf,
    const float* __restrict__ g1, const float* __restrict__ b1,
    const float* __restrict__ g2, const float* __restrict__ b2,
    __bf16* __restrict__ xs_bf) {
  const int PITCH = KDOWN + 8;
  __shared__ __attribute__((aligned(16))) __bf16 panel[16 * (KDOWN + 8)];
  __shared__ float tile[16 * CDIM];

  const int tid = threadIdx.x;
  const int n0  = blockIdx.x * 16;

  // Gather the 16-position x 1024-K patch panel into LDS.  k = ci*4+dy*2+dx:
  // (dx=0,1) is 2 contiguous bf16 in memory (wp stride 2) -> one 4B transfer.
#if USE_ASYNC
  for (int e = tid; e < 16 * (KDOWN / 2); e += 256) {
    int p = e >> 9, kk = e & (KDOWN / 2 - 1);   // kk = ci*2+dy
    int n = n0 + p;
    int b = n / LSEQ, l = n - b * LSEQ;
    int hh = l / WO, wp = l - hh * WO;
    int ci = kk >> 1, dy = kk & 1;
    const __bf16* src = x_bf + ((b * CDIM + ci) * HIN + (2 * hh + dy)) * WIN + 2 * wp;
    async_b32(src, panel + p * PITCH + kk * 2);
  }
  __builtin_amdgcn_s_wait_asynccnt(0);
  __syncthreads();
#else
  for (int e = tid; e < 16 * KDOWN; e += 256) {
    int p = e >> 10, k = e & (KDOWN - 1);
    int n = n0 + p;
    int b = n / LSEQ, l = n - b * LSEQ;
    int hh = l / WO, wp = l - hh * WO;
    int ci = k >> 2, dy = (k >> 1) & 1, dx = k & 1;
    panel[p * PITCH + k] =
        x_bf[((b * CDIM + ci) * HIN + (2 * hh + dy)) * WIN + (2 * wp + dx)];
  }
  __syncthreads();
#endif

  const int wave = tid >> 5, lane = tid & 31;
  const int half = lane >> 4, row = lane & 15;
  const int m0 = (wave * 2) * 16, m1 = m0 + 16;
  v8f acc0 = {}, acc1 = {};
  const __bf16* a0b = wdown_bf + (m0 + row) * KDOWN;
  const __bf16* a1b = wdown_bf + (m1 + row) * KDOWN;
  const __bf16* bb  = panel + row * PITCH;
  for (int kc = 0; kc < KDOWN; kc += 32) {
    __builtin_prefetch(a0b + kc + 256, 0, 1);
    v16bf bf = load_frag_k(bb + kc, half);
    v16bf a0 = load_frag_k(a0b + kc, half);
    v16bf a1 = load_frag_k(a1b + kc, half);
    acc0 = wmma_bf16(a0, bf, acc0);
    acc1 = wmma_bf16(a1, bf, acc1);
  }
  // D layout: lane -> position (N), VGPR v (+ half*8) -> channel (M)
#pragma unroll
  for (int v = 0; v < 8; ++v) {
    tile[row * CDIM + (m0 + v + half * 8)] = acc0[v];
    tile[row * CDIM + (m1 + v + half * 8)] = acc1[v];
  }
  __syncthreads();

  // Double LayerNorm over 256 channels; one 16-thread group per position.
  {
    const int g = tid >> 4;        // position 0..15 (16-lane halves of waves)
    const int t = tid & 15;
    float vals[16];
    float s = 0.f, ss = 0.f;
#pragma unroll
    for (int i = 0; i < 16; ++i) {
      float v = tile[g * CDIM + t + i * 16];
      vals[i] = v; s += v; ss += v * v;
    }
#pragma unroll
    for (int o = 8; o >= 1; o >>= 1) { s += __shfl_xor(s, o, 32); ss += __shfl_xor(ss, o, 32); }
    float mu   = s * (1.0f / CDIM);
    float rstd = rsqrtf(ss * (1.0f / CDIM) - mu * mu + 1e-5f);
    float s2 = 0.f, ss2 = 0.f;
#pragma unroll
    for (int i = 0; i < 16; ++i) {
      int c = t + i * 16;
      float y1 = (vals[i] - mu) * rstd * g1[c] + b1[c];
      vals[i] = y1; s2 += y1; ss2 += y1 * y1;
    }
#pragma unroll
    for (int o = 8; o >= 1; o >>= 1) { s2 += __shfl_xor(s2, o, 32); ss2 += __shfl_xor(ss2, o, 32); }
    float mu2   = s2 * (1.0f / CDIM);
    float rstd2 = rsqrtf(ss2 * (1.0f / CDIM) - mu2 * mu2 + 1e-5f);
    const int n = n0 + g;
#pragma unroll
    for (int i = 0; i < 16; ++i) {
      int c = t + i * 16;
      xs_bf[n * CDIM + c] = f2bf((vals[i] - mu2) * rstd2 * g2[c] + b2[c]);
    }
  }
}

// ---------------------------------------------------------------------------
// K2: in-projection GEMM (M=1024, K=256); splits xm (channel-major) and z.
// ---------------------------------------------------------------------------
__global__ __launch_bounds__(256) void k_inproj(
    const __bf16* __restrict__ xs_bf, const __bf16* __restrict__ win_bf,
    float* __restrict__ xm_f, float* __restrict__ z_f) {
  const int KT = CDIM, PITCH = CDIM + 8;
  __shared__ __attribute__((aligned(16))) __bf16 panel[16 * (CDIM + 8)];
  const int tid = threadIdx.x;
  const int n0  = blockIdx.x * 16;
  stage_panel<CDIM, 6>(xs_bf + n0 * CDIM, panel, tid);   // 256 bf16 = 128 DW rows

  const int wave = tid >> 5, lane = tid & 31;
  const int half = lane >> 4, row = lane & 15;
  v8f acc[8];
#pragma unroll
  for (int i = 0; i < 8; ++i) acc[i] = (v8f){};
  const __bf16* bb = panel + row * PITCH;
  for (int kc = 0; kc < KT; kc += 32) {
    v16bf bf = load_frag_k(bb + kc, half);
#pragma unroll
    for (int i = 0; i < 8; ++i) {
      int m = (wave * 8 + i) * 16 + row;
      v16bf af = load_frag_k(win_bf + m * KT + kc, half);
      acc[i] = wmma_bf16(af, bf, acc[i]);
    }
  }
  const int n = n0 + row;
  const int b = n / LSEQ, l = n - b * LSEQ;
#pragma unroll
  for (int i = 0; i < 8; ++i) {
#pragma unroll
    for (int v = 0; v < 8; ++v) {
      int ch = (wave * 8 + i) * 16 + v + half * 8;
      float val = acc[i][v];
      if (ch < DIN) xm_f[(b * DIN + ch) * LSEQ + l] = val;        // channel-major
      else          z_f[n * DIN + (ch - DIN)]       = val;        // row-major
    }
  }
}

// ---------------------------------------------------------------------------
// K3: depthwise causal conv1d (D_CONV=4) + SiLU; emits fp32 + bf16 xc.
// ---------------------------------------------------------------------------
__global__ __launch_bounds__(256) void k_conv_silu(
    const float* __restrict__ xm_f, const float* __restrict__ wconv,
    const float* __restrict__ bconv, float* __restrict__ xc_f,
    __bf16* __restrict__ xc_bf) {
  int idx = blockIdx.x * 256 + threadIdx.x;             // BB*DIN*LSEQ total
  int l  = idx % LSEQ;
  int bc = idx / LSEQ;
  int c  = bc % DIN;
  int b  = bc / DIN;
  const float* base = xm_f + (b * DIN + c) * LSEQ;
  float a = bconv[c];
#pragma unroll
  for (int j = 0; j < 4; ++j) {
    int li = l - 3 + j;
    if (li >= 0) a += wconv[c * 4 + j] * base[li];
  }
  float s = a / (1.0f + __expf(-a));                    // SiLU
  int n = b * LSEQ + l;
  xc_f[n * DIN + c]  = s;
  xc_bf[n * DIN + c] = f2bf(s);
}

// ---------------------------------------------------------------------------
// K4: x-projection GEMM (M=48, K=512) + dt projection (K=16) + softplus.
// ---------------------------------------------------------------------------
__global__ __launch_bounds__(256) void k_xproj_dt(
    const __bf16* __restrict__ xc_bf, const __bf16* __restrict__ wxproj_bf,
    const float* __restrict__ w_dt, const float* __restrict__ b_dt,
    float* __restrict__ dt_f, float* __restrict__ Bm_f,
    float* __restrict__ Cm_f) {
  const int KT = DIN, PITCH = DIN + 8;
  __shared__ __attribute__((aligned(16))) __bf16 panel[16 * (DIN + 8)];
  __shared__ float dbc[16 * 48];
  const int tid = threadIdx.x;
  const int n0  = blockIdx.x * 16;
  stage_panel<DIN, 7>(xc_bf + n0 * DIN, panel, tid);     // 512 bf16 = 256 DW rows

  const int wave = tid >> 5, lane = tid & 31;
  const int half = lane >> 4, row = lane & 15;
  if (wave < 3) {                       // wave-uniform branch: EXEC all-ones
    v8f acc = {};
    const __bf16* bb = panel + row * PITCH;
    const __bf16* ab = wxproj_bf + (wave * 16 + row) * KT;
    for (int kc = 0; kc < KT; kc += 32) {
      v16bf bf = load_frag_k(bb + kc, half);
      v16bf af = load_frag_k(ab + kc, half);
      acc = wmma_bf16(af, bf, acc);
    }
#pragma unroll
    for (int v = 0; v < 8; ++v)
      dbc[row * 48 + (wave * 16 + v + half * 8)] = acc[v];
  }
  __syncthreads();

  {   // B / C split
    int p = tid >> 4, s = tid & 15;
    int n = n0 + p;
    Bm_f[n * 16 + s] = dbc[p * 48 + 16 + s];
    Cm_f[n * 16 + s] = dbc[p * 48 + 32 + s];
  }
  // dt = softplus(dt_in @ w_dt.T + b_dt)  (K=16, scalar FMA is ample)
  for (int out = tid; out < 16 * DIN; out += 256) {
    int p = out & 15, c = out >> 4;
    float a = b_dt[c];
#pragma unroll
    for (int r = 0; r < DTRANK; ++r) a += dbc[p * 48 + r] * w_dt[c * DTRANK + r];
    float sp = (a > 20.0f) ? a : log1pf(__expf(a));
    dt_f[(n0 + p) * DIN + c] = sp;
  }
}

// ---------------------------------------------------------------------------
// K5: selective scan, fused with D-residual and SiLU(z) gating -> y (bf16).
// Lanes 0..15 / 16..31 each carry the 16 d_state slots of one (b,c) channel.
// ---------------------------------------------------------------------------
__global__ __launch_bounds__(256) void k_scan_gate(
    const float* __restrict__ dt_f, const float* __restrict__ Bm_f,
    const float* __restrict__ Cm_f, const float* __restrict__ xc_f,
    const float* __restrict__ z_f, const float* __restrict__ A_log,
    const float* __restrict__ Dvec, __bf16* __restrict__ y_bf) {
  const int tid  = threadIdx.x;
  const int lane = tid & 31, half = lane >> 4, s = lane & 15;
  const int wid   = blockIdx.x * 8 + (tid >> 5);
  const int combo = wid * 2 + half;           // 0..4095 = (b, c)
  const int b = combo >> 9;
  const int c = combo & (DIN - 1);
  const float As = -__expf(A_log[c * DSTATE + s]);
  const float Dc = Dvec[c];
  float h = 0.0f;
  const int nbase = b * LSEQ;
  for (int l = 0; l < LSEQ; ++l) {
    int n = nbase + l;
    float dtv = dt_f[n * DIN + c];
    float xv  = xc_f[n * DIN + c];
    float Bv  = Bm_f[n * 16 + s];
    float Cv  = Cm_f[n * 16 + s];
    float dA  = __expf(dtv * As);
    h = dA * h + dtv * Bv * xv;
    float y = h * Cv;
#pragma unroll
    for (int o = 8; o >= 1; o >>= 1) y += __shfl_xor(y, o, 32);   // 16-lane dot
    if (s == 0) {
      float zv = z_f[n * DIN + c];
      float g  = zv / (1.0f + __expf(-zv));
      y_bf[n * DIN + c] = f2bf((y + xv * Dc) * g);
    }
  }
}

// ---------------------------------------------------------------------------
// K6: out-projection GEMM (M=256, K=512) writing NCHW output directly.
// ---------------------------------------------------------------------------
__global__ __launch_bounds__(256) void k_outproj(
    const __bf16* __restrict__ y_bf, const __bf16* __restrict__ wout_bf,
    float* __restrict__ out) {
  const int KT = DIN, PITCH = DIN + 8;
  __shared__ __attribute__((aligned(16))) __bf16 panel[16 * (DIN + 8)];
  const int tid = threadIdx.x;
  const int n0  = blockIdx.x * 16;
  stage_panel<DIN, 7>(y_bf + n0 * DIN, panel, tid);

  const int wave = tid >> 5, lane = tid & 31;
  const int half = lane >> 4, row = lane & 15;
  const int m0 = (wave * 2) * 16, m1 = m0 + 16;
  v8f acc0 = {}, acc1 = {};
  const __bf16* bb  = panel + row * PITCH;
  const __bf16* a0b = wout_bf + (m0 + row) * KT;
  const __bf16* a1b = wout_bf + (m1 + row) * KT;
  for (int kc = 0; kc < KT; kc += 32) {
    __builtin_prefetch(a0b + kc + 128, 0, 1);
    v16bf bf = load_frag_k(bb + kc, half);
    v16bf a0 = load_frag_k(a0b + kc, half);
    v16bf a1 = load_frag_k(a1b + kc, half);
    acc0 = wmma_bf16(a0, bf, acc0);
    acc1 = wmma_bf16(a1, bf, acc1);
  }
  const int n = n0 + row;
  const int b = n / LSEQ, l = n - b * LSEQ;
#pragma unroll
  for (int v = 0; v < 8; ++v) {
    out[(b * CDIM + (m0 + v + half * 8)) * LSEQ + l] = acc0[v];
    out[(b * CDIM + (m1 + v + half * 8)) * LSEQ + l] = acc1[v];
  }
}

// ---------------------------------------------------------------------------
extern "C" void kernel_launch(void* const* d_in, const int* in_sizes, int n_in,
                              void* d_out, int out_size, void* d_ws, size_t ws_size,
                              hipStream_t stream) {
  (void)in_sizes; (void)n_in; (void)out_size; (void)ws_size;
  const float* x       = (const float*)d_in[0];
  const float* w_down  = (const float*)d_in[1];
  const float* g1      = (const float*)d_in[2];
  const float* b1      = (const float*)d_in[3];
  const float* g2      = (const float*)d_in[4];
  const float* b2      = (const float*)d_in[5];
  const float* w_in    = (const float*)d_in[6];
  const float* w_conv  = (const float*)d_in[7];
  const float* b_conv  = (const float*)d_in[8];
  const float* w_xproj = (const float*)d_in[9];
  const float* w_dt    = (const float*)d_in[10];
  const float* b_dt    = (const float*)d_in[11];
  const float* A_log   = (const float*)d_in[12];
  const float* Dvec    = (const float*)d_in[13];
  const float* w_out   = (const float*)d_in[14];
  float* out = (float*)d_out;

  char* ws = (char*)d_ws;
  size_t off = 0;
  auto carve = [&](size_t bytes) {
    char* p = ws + off;
    off = (off + bytes + 255) & ~(size_t)255;
    return p;
  };
  __bf16* wdown_bf  = (__bf16*)carve((size_t)CDIM * KDOWN * 2);
  __bf16* win_bf    = (__bf16*)carve((size_t)2 * DIN * CDIM * 2);
  __bf16* wxproj_bf = (__bf16*)carve((size_t)48 * DIN * 2);
  __bf16* wout_bf   = (__bf16*)carve((size_t)CDIM * DIN * 2);
  __bf16* x_bf      = (__bf16*)carve((size_t)BB * CDIM * HIN * WIN * 2);
  __bf16* xs_bf     = (__bf16*)carve((size_t)NPOS * CDIM * 2);
  float*  xm_f      = (float*) carve((size_t)BB * DIN * LSEQ * 4);
  float*  z_f       = (float*) carve((size_t)NPOS * DIN * 4);
  float*  xc_f      = (float*) carve((size_t)NPOS * DIN * 4);
  __bf16* xc_bf     = (__bf16*)carve((size_t)NPOS * DIN * 2);
  float*  dt_f      = (float*) carve((size_t)NPOS * DIN * 4);
  float*  Bm_f      = (float*) carve((size_t)NPOS * 16 * 4);
  float*  Cm_f      = (float*) carve((size_t)NPOS * 16 * 4);
  __bf16* y_bf      = (__bf16*)carve((size_t)NPOS * DIN * 2);

  const int ncvt = CDIM * KDOWN + 2 * DIN * CDIM + 48 * DIN + CDIM * DIN;
  k_convert<<<(ncvt + 255) / 256, 256, 0, stream>>>(
      w_down, w_in, w_xproj, w_out, wdown_bf, win_bf, wxproj_bf, wout_bf);

  k_convert_x<<<(BB * CDIM * HIN * WIN) / 256, 256, 0, stream>>>(x, x_bf);

  k_merge_ln<<<NPOS / 16, 256, 0, stream>>>(x_bf, wdown_bf, g1, b1, g2, b2, xs_bf);

  k_inproj<<<NPOS / 16, 256, 0, stream>>>(xs_bf, win_bf, xm_f, z_f);

  k_conv_silu<<<(BB * DIN * LSEQ) / 256, 256, 0, stream>>>(
      xm_f, w_conv, b_conv, xc_f, xc_bf);

  k_xproj_dt<<<NPOS / 16, 256, 0, stream>>>(
      xc_bf, wxproj_bf, w_dt, b_dt, dt_f, Bm_f, Cm_f);

  k_scan_gate<<<256, 256, 0, stream>>>(
      dt_f, Bm_f, Cm_f, xc_f, z_f, A_log, Dvec, y_bf);

  k_outproj<<<NPOS / 16, 256, 0, stream>>>(y_bf, wout_bf, out);
}
